// DGCNNAgent_70377334112594
// MI455X (gfx1250) — matrix operand order repaired and checked by reading, not compiled, not run on hardware
//
#include <hip/hip_runtime.h>
#include <hip/hip_bf16.h>
#include <stdint.h>

// ---------------------------------------------------------------------------
// DGCNN head for MI455X (gfx1250, wave32, WMMA + TDM).
// GEMM layers use V_WMMA_F32_16X16X32_F16 (f16 inputs, f32 accum) — activations
// are BN-normalized so f16 operands are safe; accumulation stays f32.
// Graph-feature gather is fused into WMMA B-operand construction from LDS.
// Edge-conv x-tiles are staged into LDS via the Tensor Data Mover.
// ---------------------------------------------------------------------------

typedef __attribute__((ext_vector_type(16))) _Float16 v16h;
typedef __attribute__((ext_vector_type(8)))  float    v8f;
typedef __attribute__((ext_vector_type(4)))  unsigned int u32x4;
typedef __attribute__((ext_vector_type(8)))  int      i32x8;
typedef __attribute__((ext_vector_type(4)))  int      i32x4;

#define NB   2048   // batch
#define NP   100    // points
#define KNN  10

__device__ __forceinline__ float lrelu(float v) { return v >= 0.f ? v : 0.2f * v; }

// ----------------------------- utility kernels -----------------------------

__global__ void k_zero(float* __restrict__ p, int n) {
  int i = blockIdx.x * blockDim.x + threadIdx.x;
  if (i < n) p[i] = 0.f;
}

// conv0: obs[b,c,n] = in[b, n*6+c]; y[b,o,n] = sum_c w0[o,c]*obs  (raw, pre-BN)
__global__ void k_conv0(const float* __restrict__ in, const float* __restrict__ w0,
                        float* __restrict__ x0) {
  __shared__ float srow[600];
  __shared__ float sw[32 * 6];
  const int b = blockIdx.x, t = threadIdx.x;
  for (int i = t; i < 600; i += 128) srow[i] = in[(size_t)b * 600 + i];
  for (int i = t; i < 192; i += 128) sw[i] = w0[i];
  __syncthreads();
  for (int i = t; i < 32 * NP; i += 128) {
    int o = i / NP, n = i % NP;
    float s = 0.f;
#pragma unroll
    for (int c = 0; c < 6; ++c) s += sw[o * 6 + c] * srow[n * 6 + c];
    x0[(size_t)b * 32 * NP + i] = s;
  }
}

// per-channel sum/sumsq over (B, C, S): one block per (b,c) row.
__global__ void k_stats(const float* __restrict__ y, float* __restrict__ sums,
                        int C, int S) {
  __shared__ float rs[128], rq[128];
  const int bc = blockIdx.x;
  const int c = bc % C;
  const float* row = y + (size_t)bc * S;
  float s = 0.f, q = 0.f;
  for (int i = threadIdx.x; i < S; i += 128) { float v = row[i]; s += v; q += v * v; }
  rs[threadIdx.x] = s; rq[threadIdx.x] = q;
  __syncthreads();
  for (int off = 64; off > 0; off >>= 1) {
    if (threadIdx.x < off) {
      rs[threadIdx.x] += rs[threadIdx.x + off];
      rq[threadIdx.x] += rq[threadIdx.x + off];
    }
    __syncthreads();
  }
  if (threadIdx.x == 0) {
    atomicAdd(&sums[c], rs[0]);
    atomicAdd(&sums[C + c], rq[0]);
  }
}

// ab[c]=g/sqrt(var+eps); ab[C+c]=beta - ab[c]*mean
__global__ void k_affine(const float* __restrict__ sums, const float* __restrict__ g,
                         const float* __restrict__ beta, float* __restrict__ ab,
                         int C, float invcount) {
  int c = blockIdx.x * blockDim.x + threadIdx.x;
  if (c < C) {
    float mean = sums[c] * invcount;
    float var  = sums[C + c] * invcount - mean * mean;
    float a = g[c] * rsqrtf(var + 1e-5f);
    ab[c] = a;
    ab[C + c] = beta[c] - a * mean;
  }
}

// in-place x = lrelu(a*x + b) over (B, C, S)
__global__ void k_apply(float* __restrict__ x, const float* __restrict__ ab,
                        int C, int S, int total) {
  int i = blockIdx.x * blockDim.x + threadIdx.x;
  if (i >= total) return;
  int c = (i / S) % C;
  x[i] = lrelu(ab[c] * x[i] + ab[C + c]);
}

// kNN: per-batch 100x100 neg-sq-distances, top-10 indices (ties -> lower index)
__global__ void k_knn(const float* __restrict__ x, int* __restrict__ idx) {
  __shared__ float xs[32 * NP];
  const int b = blockIdx.x, t = threadIdx.x;
  const float* xb = x + (size_t)b * 32 * NP;
  for (int i = t; i < 32 * NP; i += 128) xs[i] = xb[i];
  __syncthreads();
  if (t >= NP) return;
  float vals[KNN];
  int   inds[KNN];
#pragma unroll
  for (int j = 0; j < KNN; ++j) { vals[j] = -3.4e38f; inds[j] = 0; }
  for (int m = 0; m < NP; ++m) {
    float d = 0.f;
#pragma unroll
    for (int c = 0; c < 32; ++c) {
      float df = xs[c * NP + t] - xs[c * NP + m];
      d += df * df;
    }
    d = -d;  // pd = -||xn - xm||^2, top_k descending
    if (d > vals[KNN - 1]) {
      vals[KNN - 1] = d; inds[KNN - 1] = m;
#pragma unroll
      for (int j = KNN - 1; j > 0; --j) {
        if (vals[j] > vals[j - 1]) {
          float tv = vals[j]; vals[j] = vals[j - 1]; vals[j - 1] = tv;
          int   ti = inds[j]; inds[j] = inds[j - 1]; inds[j - 1] = ti;
        }
      }
    }
  }
#pragma unroll
  for (int j = 0; j < KNN; ++j)
    idx[(size_t)b * NP * KNN + t * KNN + j] = inds[j];
}

// ---------------------- TDM staging helper (gfx1250) -----------------------
// Issue a 1-row TENSOR_LOAD_TO_LDS of `elems` f32 from `src` to LDS offset
// `lds_off` (bytes). D# layout per CDNA5 ISA §8 (group0/group1, type=2).
// This toolchain exposes the 6-arg builtin: (g0, g1, g2, g3, g4, cpol).
__device__ __forceinline__ void tdm_load_f32(const float* src, unsigned int lds_off,
                                             unsigned int elems) {
#if defined(__gfx1250__) && __has_builtin(__builtin_amdgcn_tensor_load_to_lds)
  unsigned long long ga = (unsigned long long)(uintptr_t)src;
  u32x4 g0;
  g0[0] = 1u;                                        // count=1 valid descriptor
  g0[1] = lds_off;                                   // lds_addr (bytes)
  g0[2] = (unsigned int)ga;                          // global_addr[31:0]
  g0[3] = (unsigned int)((ga >> 32) & 0x01FFFFFFu)   // global_addr[56:32]
        | 0x80000000u;                               // type=2 ("image")
  i32x8 g1;
  g1[0] = 0x00020000;                 // workgroup_mask=0, data_size=2 (4B)
  g1[1] = (int)((elems & 0xFFFFu) << 16);            // tensor_dim0[15:0]
  g1[2] = 0x00010000 | (int)(elems >> 16);           // dim0[31:16] | tensor_dim1=1
  g1[3] = (int)((elems & 0xFFFFu) << 16);            // tile_dim0 = elems
  g1[4] = 1;                                         // tile_dim1=1
  g1[5] = (int)elems;                                // tensor_dim0_stride
  g1[6] = 0; g1[7] = 0;
  i32x4 g2 = {0, 0, 0, 0}, g3 = {0, 0, 0, 0};        // 2D tensor: groups 2/3 unused
  i32x8 g4 = {0, 0, 0, 0, 0, 0, 0, 0};               // extra group (clang-23 form)
  __builtin_amdgcn_tensor_load_to_lds(g0, g1, g2, g3, g4, 0);
  __builtin_amdgcn_s_wait_tensorcnt(0);
#else
  (void)src; (void)lds_off; (void)elems;
#endif
}

// ------------------------- WMMA GEMM kernels -------------------------------
// Operand packing used here (wave32; contiguous k, not the ISA table order):
//   A (16x32 f16): lane l (m=l%16, h=l/16), element e: k = e + 16h
//   B (32x16 f16): lane l (n=l%16, h=l/16), element e: k = e + 16h
//   D (16x16 f32): lane l, vgpr r: m = r + 8h, n = l%16

// Edge conv: y[b,o,p] = sum_{c<32} W[o,c]*(x[c,idx[p]]-x[c,n]) + sum W[o,32+c]*x[c,n]
// x: (B,32,100) activated; W: (32,64) row-major; y: (B,32,1000) raw pre-BN.
__global__ void k_edgeconv_wmma(const float* __restrict__ x, const int* __restrict__ idx,
                                const float* __restrict__ W, float* __restrict__ y) {
  __shared__ float xs[32 * NP];
  const int b = blockIdx.x;
  const float* xb = x + (size_t)b * 32 * NP;
#if defined(__gfx1250__) && __has_builtin(__builtin_amdgcn_tensor_load_to_lds)
  if (threadIdx.x < 32)  // one wave issues the TDM copy and waits TENSORcnt
    tdm_load_f32(xb, (unsigned int)(uintptr_t)xs, 32 * NP);
#else
  for (int i = threadIdx.x; i < 32 * NP; i += 128) xs[i] = xb[i];
#endif
  __syncthreads();
  const int lane = threadIdx.x & 31;
  const int wave = threadIdx.x >> 5;
  const int h = lane >> 4, ln = lane & 15;
  const int* idxb = idx + (size_t)b * 1000;
  const int S = 1000, NT = 63;  // ceil(1000/16)

  // A operands are tile-invariant: pack once per wave (4 x v16h held in VGPRs).
  v16h a00, a01, a10, a11;  // a{mtile}{kstep}
  {
    const float* w0r = W + (size_t)(ln)      * 64 + 16 * h;
    const float* w1r = W + (size_t)(16 + ln) * 64 + 16 * h;
#pragma unroll
    for (int e = 0; e < 16; ++e) {
      int k = e;
      a00[e] = (_Float16)w0r[k];
      a01[e] = (_Float16)w0r[32 + k];
      a10[e] = (_Float16)w1r[k];
      a11[e] = (_Float16)w1r[32 + k];
    }
  }
  for (int t = wave; t < NT; t += 4) {            // uniform per wave -> EXEC all-1s
    const int p = t * 16 + ln;
    const int pc = p < S ? p : S - 1;             // clamp loads, mask stores
    const int n = pc / KNN;
    const int j = idxb[pc];
    v8f acc0 = {}, acc1 = {};
    v16h bmat;
#pragma unroll
    for (int e = 0; e < 16; ++e) {                // kstep 0: gathered - center
      int c = e + 16 * h;
      bmat[e] = (_Float16)(xs[c * NP + j] - xs[c * NP + n]);
    }
    acc0 = __builtin_amdgcn_wmma_f32_16x16x32_f16(false, a00, false, bmat,
                                                  (short)0, acc0, false, false);
    acc1 = __builtin_amdgcn_wmma_f32_16x16x32_f16(false, a10, false, bmat,
                                                  (short)0, acc1, false, false);
#pragma unroll
    for (int e = 0; e < 16; ++e) {                // kstep 1: center features
      int c = e + 16 * h;
      bmat[e] = (_Float16)xs[c * NP + n];
    }
    acc0 = __builtin_amdgcn_wmma_f32_16x16x32_f16(false, a01, false, bmat,
                                                  (short)0, acc0, false, false);
    acc1 = __builtin_amdgcn_wmma_f32_16x16x32_f16(false, a11, false, bmat,
                                                  (short)0, acc1, false, false);
    if (p < S) {
#pragma unroll
      for (int r = 0; r < 8; ++r) {
        int m0 = r + 8 * h;
        y[((size_t)b * 32 + m0)      * S + p] = acc0[r];
        y[((size_t)b * 32 + 16 + m0) * S + p] = acc1[r];
      }
    }
  }
}

// Generic batched GEMM: out[b,o,s] = bias?[b,o] + sum_c W[o*ldw+wofs+c]*act(in[b,c,s])
// act = lrelu(ab[c]*v + ab[Cin+c]) when ab != nullptr, else identity.
// Input slab is activated + converted to f16 in dynamic LDS once per block.
__global__ void k_conv_wmma(const float* __restrict__ in, const float* __restrict__ ab,
                            const float* __restrict__ W, int ldw, int wofs,
                            const float* __restrict__ bias, float* __restrict__ out,
                            int Cin, int Cout, int S) {
  extern __shared__ _Float16 sh[];   // Cin*S activated f16
  const int b = blockIdx.x;
  const float* inb = in + (size_t)b * Cin * S;
  for (int i = threadIdx.x; i < Cin * S; i += 128) {
    float v = inb[i];
    if (ab) { int c = i / S; v = lrelu(ab[c] * v + ab[Cin + c]); }
    sh[i] = (_Float16)v;
  }
  __syncthreads();
  const int lane = threadIdx.x & 31;
  const int wave = threadIdx.x >> 5;
  const int h = lane >> 4, ln = lane & 15;
  const int mtiles = Cout >> 4;
  const int ntiles = (S + 15) >> 4;
  const int ksteps = Cin >> 5;                    // Cin in {32,96,128}
  for (int mt = 0; mt < mtiles; ++mt) {
    if (mt + 1 < mtiles)                          // prefetch next W M-tile rows
      __builtin_prefetch(W + (size_t)((mt + 1) * 16 + ln) * ldw + wofs, 0, 0);
    v16h areg[4];                                 // A held across all N-tiles
#pragma unroll
    for (int ks = 0; ks < 4; ++ks) {
      if (ks < ksteps) {
        const float* wrow = W + (size_t)(mt * 16 + ln) * ldw + wofs + ks * 32 + 16 * h;
#pragma unroll
        for (int e = 0; e < 16; ++e) {
          int k = e;
          areg[ks][e] = (_Float16)wrow[k];
        }
      }
    }
    for (int nt = wave; nt < ntiles; nt += 4) {   // N-tiles across the 4 waves
      const int s = nt * 16 + ln;
      const int sc = s < S ? s : S - 1;
      v8f acc = {};
#pragma unroll
      for (int ks = 0; ks < 4; ++ks) {
        if (ks < ksteps) {
          v16h bmat;
#pragma unroll
          for (int e = 0; e < 16; ++e)
            bmat[e] = sh[(ks * 32 + e + 16 * h) * S + sc];
          acc = __builtin_amdgcn_wmma_f32_16x16x32_f16(false, areg[ks], false, bmat,
                                                       (short)0, acc, false, false);
        }
      }
      if (s < S) {
#pragma unroll
        for (int r = 0; r < 8; ++r) {
          int m = mt * 16 + r + 8 * h;
          float v = acc[r];
          if (bias) v += bias[(size_t)b * Cout + m];
          out[((size_t)b * Cout + m) * S + s] = v;
        }
      }
    }
  }
}

// ------------------------- reductions / glue -------------------------------

// x[b,o,n] = max_k lrelu(ab applied y[b,o,n*10+k]);  y: (B,32,1000)
__global__ void k_max_k(const float* __restrict__ y, const float* __restrict__ ab,
                        float* __restrict__ x, int total) {
  int i = blockIdx.x * blockDim.x + threadIdx.x;
  if (i >= total) return;
  int c = (i / NP) % 32;
  float a = ab[c], bb = ab[32 + c];
  const float* p = y + (size_t)(i / NP) * 1000 + (size_t)(i % NP) * KNN;
  float m = -3.4e38f;
#pragma unroll
  for (int k = 0; k < KNN; ++k) {
    float v = lrelu(a * p[k] + bb);
    m = v > m ? v : m;
  }
  x[i] = m;
}

// cat96[b,c,n]: c<32 -> x1, c<64 -> x2, else x3
__global__ void k_cat3(const float* __restrict__ x1, const float* __restrict__ x2,
                       const float* __restrict__ x3, float* __restrict__ cat, int total) {
  int i = blockIdx.x * blockDim.x + threadIdx.x;
  if (i >= total) return;
  int b = i / (96 * NP);
  int c = (i / NP) % 96;
  int n = i % NP;
  const float* src = (c < 32) ? x1 : (c < 64) ? x2 : x3;
  cat[i] = src[((size_t)b * 32 + (c & 31)) * NP + n];
}

// xg[b,o] = max_n lrelu(ab applied y6[b,o,n])
__global__ void k_max_n(const float* __restrict__ y, const float* __restrict__ ab,
                        float* __restrict__ xg, int total) {
  int i = blockIdx.x * blockDim.x + threadIdx.x;
  if (i >= total) return;
  int c = i % 512;
  float a = ab[c], bb = ab[512 + c];
  const float* p = y + (size_t)i * NP;
  float m = -3.4e38f;
  for (int n = 0; n < NP; ++n) {
    float v = lrelu(a * p[n] + bb);
    m = v > m ? v : m;
  }
  xg[i] = m;
}

// bias7[b,o] = sum_{c<512} w7[o*608+c] * xg[b,c]
__global__ void k_matvec(const float* __restrict__ xg, const float* __restrict__ w7,
                         float* __restrict__ bias7) {
  __shared__ float sg[512];
  const int b = blockIdx.x, t = threadIdx.x;
  for (int i = t; i < 512; i += 128) sg[i] = xg[(size_t)b * 512 + i];
  __syncthreads();
  float s = 0.f;
  for (int c = 0; c < 512; ++c) s += w7[(size_t)t * 608 + c] * sg[c];
  bias7[(size_t)b * 128 + t] = s;
}

// logits[n] = sum_c w9[c]*lrelu(ab applied y8[b,c,n]); softmax over n -> out[b,n]
__global__ void k_softmax(const float* __restrict__ y8, const float* __restrict__ ab,
                          const float* __restrict__ w9, float* __restrict__ out) {
  __shared__ float logit[NP];
  __shared__ float red[128];
  const int b = blockIdx.x, t = threadIdx.x;
  if (t < NP) {
    float s = 0.f;
#pragma unroll
    for (int c = 0; c < 32; ++c) {
      float v = lrelu(ab[c] * y8[((size_t)b * 32 + c) * NP + t] + ab[32 + c]);
      s += w9[c] * v;
    }
    logit[t] = s;
  }
  __syncthreads();
  red[t] = (t < NP) ? logit[t] : -3.4e38f;
  __syncthreads();
  for (int off = 64; off > 0; off >>= 1) {
    if (t < off) red[t] = red[t] > red[t + off] ? red[t] : red[t + off];
    __syncthreads();
  }
  const float mx = red[0];
  __syncthreads();
  float e = (t < NP) ? __expf(logit[t] - mx) : 0.f;
  red[t] = e;
  __syncthreads();
  for (int off = 64; off > 0; off >>= 1) {
    if (t < off) red[t] += red[t + off];
    __syncthreads();
  }
  if (t < NP) out[(size_t)b * NP + t] = e / red[0];
}

// ------------------------------- host side ---------------------------------

extern "C" void kernel_launch(void* const* d_in, const int* in_sizes, int n_in,
                              void* d_out, int out_size, void* d_ws, size_t ws_size,
                              hipStream_t stream) {
  (void)in_sizes; (void)n_in; (void)out_size; (void)ws_size;
  // d_in order: inputs, w0..w9, then g0,b0,g1,b1,...,g8,b8
  const float* inp = (const float*)d_in[0];
  const float* w[10];
  for (int i = 0; i < 10; ++i) w[i] = (const float*)d_in[1 + i];
  const float* g[9]; const float* bt[9];
  for (int i = 0; i < 9; ++i) { g[i] = (const float*)d_in[11 + 2 * i]; bt[i] = (const float*)d_in[12 + 2 * i]; }

  float* ws = (float*)d_ws;
  const size_t B = NB;
  // workspace layout (floats) — requires ws_size >= ~670 MB
  const size_t f_x0  = 0;                              // B*3200
  const size_t f_x1  = f_x0  + B * 3200;
  const size_t f_x2  = f_x1  + B * 3200;
  const size_t f_x3  = f_x2  + B * 3200;
  const size_t f_cat = f_x3  + B * 3200;               // B*9600
  const size_t f_idx = f_cat + B * 9600;               // B*1000 ints
  const size_t f_yA  = f_idx + B * 1000;               // B*32000
  const size_t f_yB  = f_yA  + B * 32000;              // B*32000
  const size_t f_xg  = f_yB  + B * 32000;              // B*512
  const size_t f_b7  = f_xg  + B * 512;                // B*128
  const size_t f_sum = f_b7  + B * 128;                // 1024
  const size_t f_ab0 = f_sum + 1024;                   // 64
  const size_t f_abA = f_ab0 + 64;
  const size_t f_abB = f_abA + 64;
  const size_t f_ab6 = f_abB + 64;                     // 1024
  const size_t f_ab7 = f_ab6 + 1024;                   // 256
  const size_t f_ab8 = f_ab7 + 256;                    // 64
  // overlays (y6/y7/y8 reuse the yA/yB region; lifetimes are disjoint)
  float* y6 = ws + f_yA;                               // B*51200 <= B*64000
  float* y7 = ws + f_yA;                               // B*12800
  float* y8 = ws + f_yA + B * 12800;                   // B*3200
  int* idxbuf = (int*)(ws + f_idx);

  auto stats_affine = [&](const float* y, int C, int S, const float* gg,
                          const float* bb, float* abdst) {
    k_zero<<<(2 * C + 255) / 256, 256, 0, stream>>>(ws + f_sum, 2 * C);
    k_stats<<<dim3((unsigned)(B * C)), 128, 0, stream>>>(y, ws + f_sum, C, S);
    k_affine<<<(C + 255) / 256, 256, 0, stream>>>(ws + f_sum, gg, bb, abdst, C,
                                                  1.0f / (float)(B * (size_t)S));
  };

  // ---- conv0 + BN0 + lrelu (in place) ----
  k_conv0<<<NB, 128, 0, stream>>>(inp, w[0], ws + f_x0);
  stats_affine(ws + f_x0, 32, NP, g[0], bt[0], ws + f_ab0);
  k_apply<<<(int)((B * 3200 + 255) / 256), 256, 0, stream>>>(ws + f_x0, ws + f_ab0,
                                                             32, NP, (int)(B * 3200));

  // ---- edge blocks ----
  struct Edge { size_t src; const float* wa; const float* wb; int gi_a, gi_b; size_t dst; };
  const Edge edges[3] = {
      {f_x0, w[1], w[2], 1, 2, f_x1},
      {f_x1, w[3], w[4], 3, 4, f_x2},
      {f_x2, w[5], nullptr, 5, -1, f_x3},
  };
  for (int e = 0; e < 3; ++e) {
    const Edge& E = edges[e];
    k_knn<<<NB, 128, 0, stream>>>(ws + E.src, idxbuf);
    k_edgeconv_wmma<<<NB, 128, 0, stream>>>(ws + E.src, idxbuf, E.wa, ws + f_yA);
    stats_affine(ws + f_yA, 32, 1000, g[E.gi_a], bt[E.gi_a], ws + f_abA);
    if (E.wb) {
      k_conv_wmma<<<NB, 128, 32 * 1000 * sizeof(_Float16), stream>>>(
          ws + f_yA, ws + f_abA, E.wb, 32, 0, nullptr, ws + f_yB, 32, 32, 1000);
      stats_affine(ws + f_yB, 32, 1000, g[E.gi_b], bt[E.gi_b], ws + f_abB);
      k_max_k<<<(int)((B * 3200 + 255) / 256), 256, 0, stream>>>(
          ws + f_yB, ws + f_abB, ws + E.dst, (int)(B * 3200));
    } else {
      k_max_k<<<(int)((B * 3200 + 255) / 256), 256, 0, stream>>>(
          ws + f_yA, ws + f_abA, ws + E.dst, (int)(B * 3200));
    }
  }

  // ---- cat96 = [x1;x2;x3] ----
  k_cat3<<<(int)((B * 9600 + 255) / 256), 256, 0, stream>>>(
      ws + f_x1, ws + f_x2, ws + f_x3, ws + f_cat, (int)(B * 9600));

  // ---- w6: (512x96) GEMM, BN6, xg = max_n ----
  k_conv_wmma<<<NB, 128, 96 * NP * sizeof(_Float16), stream>>>(
      ws + f_cat, nullptr, w[6], 96, 0, nullptr, y6, 96, 512, NP);
  stats_affine(y6, 512, NP, g[6], bt[6], ws + f_ab6);
  k_max_n<<<(int)((B * 512 + 255) / 256), 256, 0, stream>>>(y6, ws + f_ab6,
                                                            ws + f_xg, (int)(B * 512));

  // ---- w7: fold broadcast xg into per-(b,o) bias, GEMM over cat96 cols ----
  k_matvec<<<NB, 128, 0, stream>>>(ws + f_xg, w[7], ws + f_b7);
  k_conv_wmma<<<NB, 128, 96 * NP * sizeof(_Float16), stream>>>(
      ws + f_cat, nullptr, w[7], 608, 512, ws + f_b7, y7, 96, 128, NP);
  stats_affine(y7, 128, NP, g[7], bt[7], ws + f_ab7);

  // ---- w8: (32x128) GEMM with BN7+lrelu fused on input ----
  k_conv_wmma<<<NB, 128, 128 * NP * sizeof(_Float16), stream>>>(
      y7, ws + f_ab7, w[8], 128, 0, nullptr, y8, 128, 32, NP);
  stats_affine(y8, 32, NP, g[8], bt[8], ws + f_ab8);

  // ---- w9 dot + softmax over points ----
  k_softmax<<<NB, 128, 0, stream>>>(y8, ws + f_ab8, w[9], (float*)d_out);
}
